// ChannelPredictor_56925496541808
// MI455X (gfx1250) — matrix operand order repaired and checked by reading, not compiled, not run on hardware
//
#include <hip/hip_runtime.h>
#include <hip/hip_bf16.h>

// ---------------------------------------------------------------------------
// ChannelPredictor for MI455X (gfx1250, wave32, WMMA bf16, async-to-LDS)
//   b=16, NC=4, NV=512, D=512, N=t*h*w=1024, TOK=b*N=16384
// ---------------------------------------------------------------------------

#define Dn   512
#define NVn  512
#define NCn  4
#define Nn   1024
#define TOKS 16384

typedef __attribute__((ext_vector_type(16))) __bf16 v16bf;
typedef __attribute__((ext_vector_type(8)))  float  v8f;

union Frag {
  v16bf v;
  uint4 q[2];
};

__device__ __forceinline__ unsigned short f2bf(float f) {
  unsigned int u = __float_as_uint(f);
  u += 0x7FFFu + ((u >> 16) & 1u);          // round-to-nearest-even
  return (unsigned short)(u >> 16);
}

// CDNA5 async copy: 16B global -> LDS per lane, tracked on ASYNCcnt.
__device__ __forceinline__ void async_ld16(unsigned ldsOff, const void* g) {
  asm volatile("global_load_async_to_lds_b128 %0, %1, off"
               :: "v"(ldsOff), "v"((unsigned long long)g) : "memory");
}
__device__ __forceinline__ void async_wait0() {
  asm volatile("s_wait_asynccnt 0x0" ::: "memory");
}

// ---------------------------------------------------------------------------
// Kernel 1: LayerNorm over D + cast to bf16, token-major output Xbf[tok][d].
// ---------------------------------------------------------------------------
#define LN_PITCH 516   // f32 pitch; avoids 64-bank clash on column writes

__global__ void ln_cast_kernel(const float* __restrict__ yl,
                               const float* __restrict__ gamma,
                               const float* __restrict__ beta,
                               unsigned short* __restrict__ xbf) {
  extern __shared__ float tile[];                     // [64][LN_PITCH]
  const int tok0 = blockIdx.x * 64;
  const int bb = tok0 >> 10;
  const int n0 = tok0 & (Nn - 1);
  const float* src = yl + (size_t)bb * Dn * Nn + n0;

  for (int it = 0; it < (64 * 512) / 256; ++it) {
    int flat = threadIdx.x + it * 256;
    int n = flat & 63;
    int d = flat >> 6;
    tile[n * LN_PITCH + d] = src[(size_t)d * Nn + n];
  }
  __syncthreads();

  const int lane = threadIdx.x & 31;
  const int wv = threadIdx.x >> 5;
  for (int t = wv * 8; t < wv * 8 + 8; ++t) {
    const float* row = tile + t * LN_PITCH;
    float s = 0.f, ss = 0.f;
    float v[16];
#pragma unroll
    for (int j = 0; j < 16; ++j) {
      float x = row[lane + 32 * j];
      v[j] = x; s += x; ss += x * x;
    }
#pragma unroll
    for (int m = 16; m >= 1; m >>= 1) {
      s  += __shfl_xor(s,  m, 32);
      ss += __shfl_xor(ss, m, 32);
    }
    float mu  = s * (1.f / 512.f);
    float var = ss * (1.f / 512.f) - mu * mu;
    float inv = rsqrtf(var + 1e-5f);
    unsigned short* dst = xbf + (size_t)(tok0 + t) * Dn;
#pragma unroll
    for (int j = 0; j < 16; ++j) {
      int d = lane + 32 * j;
      dst[d] = f2bf((v[j] - mu) * inv * gamma[d] + beta[d]);
    }
  }
}

// ---------------------------------------------------------------------------
// Kernel 2: transpose one-hot column blocks: Gt[p][v][d] = Uk_w[d][512+512k'+v]
// ---------------------------------------------------------------------------
__global__ void gather_transpose_kernel(const float* __restrict__ U1,
                                        const float* __restrict__ U2,
                                        const float* __restrict__ U3,
                                        float* __restrict__ Gt) {
  int flat = blockIdx.x * blockDim.x + threadIdx.x;
  if (flat >= 6 * 512 * 512) return;
  int p = flat >> 18;
  int v = (flat >> 9) & 511;
  int d = flat & 511;
  const float* src; int rowlen; int kp;
  if (p == 0)      { src = U1; rowlen = 1024; kp = 0; }
  else if (p == 1) { src = U2; rowlen = 1536; kp = 0; }
  else if (p == 2) { src = U2; rowlen = 1536; kp = 1; }
  else if (p == 3) { src = U3; rowlen = 2048; kp = 0; }
  else if (p == 4) { src = U3; rowlen = 2048; kp = 1; }
  else             { src = U3; rowlen = 2048; kp = 2; }
  Gt[flat] = src[(size_t)d * rowlen + 512 + kp * 512 + v];
}

// ---------------------------------------------------------------------------
// Kernel 3: bf16-cast dense weight parts (B-operand [N][K] layouts).
// ---------------------------------------------------------------------------
__global__ void wcast_kernel(const float* __restrict__ U0, const float* __restrict__ U1,
                             const float* __restrict__ U2, const float* __restrict__ U3,
                             const float* __restrict__ Pw,
                             unsigned short* __restrict__ Wbf,
                             unsigned short* __restrict__ Pbf) {
  int flat = blockIdx.x * blockDim.x + threadIdx.x;
  if (flat < 4 * 512 * 512) {
    int k = flat >> 18;
    int r = flat & (512 * 512 - 1);
    int d = r >> 9, kc = r & 511;
    const float* U = (k == 0) ? U0 : (k == 1) ? U1 : (k == 2) ? U2 : U3;
    Wbf[flat] = f2bf(U[(size_t)d * (512 * (k + 1)) + kc]);
  } else if (flat < 5 * 512 * 512) {
    int r = flat - 4 * 512 * 512;
    Pbf[r] = f2bf(Pw[r]);
  }
}

// ---------------------------------------------------------------------------
// Kernel 4: fused two-stage WMMA GEMM per (32-token tile, channel k).
//   Stage 1: U[32][512] = X[32][512] @ Wkᵀ + (bias + one-hot gathers)
//   Stage 2: O[32][512] = relu(U) @ P_wᵀ + P_b, stored transposed (b,NV,n).
// Block = 256 thr = 8 waves; wave w owns output columns [64w, 64w+64);
// per wave: 2x4 accum tiles (64 VGPRs) -> no vgpr_msb pressure.
// Global->LDS streaming uses global_load_async_to_lds_b128 (ASYNCcnt).
// ---------------------------------------------------------------------------
#define XS_PITCH 40     // shorts (80B rows: 16B-aligned b128, conflict-free)
#define WS_PITCH 40
#define US_PITCH 520
#define BA_PITCH 516

// byte offsets inside the (purely dynamic) LDS segment
#define BA_OFF 0                         // 32*516*4 = 66048
#define US_OFF 66048                     // 32*520*2 = 33280
#define XS_OFF (US_OFF + 33280)          // 2*32*40*2 = 5120   -> 99328
#define WS_OFF (XS_OFF + 5120)           // 2*512*40*2 = 81920 -> 104448
#define SM_TOT (WS_OFF + 81920)          // 186368

__global__ __launch_bounds__(256)
void fused_gemm_kernel(const unsigned short* __restrict__ Xbf,
                       const unsigned short* __restrict__ Wbf,
                       const unsigned short* __restrict__ Pbf,
                       const float* __restrict__ Gt,
                       const int* __restrict__ sidx,
                       const float* __restrict__ Ub0, const float* __restrict__ Ub1,
                       const float* __restrict__ Ub2, const float* __restrict__ Ub3,
                       const float* __restrict__ Pb,
                       float* __restrict__ out) {
  extern __shared__ char smem[];
  float*          Bacc = (float*)(smem + BA_OFF);           // [32][BA_PITCH]
  unsigned short* Us   = (unsigned short*)(smem + US_OFF);  // [32][US_PITCH]
  unsigned short* Xs   = (unsigned short*)(smem + XS_OFF);  // 2x[32][XS_PITCH]
  unsigned short* Ws   = (unsigned short*)(smem + WS_OFF);  // 2x[512][WS_PITCH]

  const int k    = blockIdx.y;
  const int tok0 = blockIdx.x * 32;
  const int bb   = tok0 >> 10;
  const int n0   = tok0 & (Nn - 1);
  const int tid  = threadIdx.x;
  const int lane = tid & 31;
  const int wv   = tid >> 5;
  const int l16  = lane & 15;
  const int lhalf = lane >> 4;

  // ---- bias + one-hot gather tile: Bacc[i][d] ----
  const float* Ub = (k == 0) ? Ub0 : (k == 1) ? Ub1 : (k == 2) ? Ub2 : Ub3;
  const int pbase = (k <= 1) ? 0 : (k == 2) ? 1 : 3;
  for (int i = 0; i < 32; ++i) {
    for (int d = tid; d < 512; d += 256) {
      float acc = Ub[d];
      for (int kp = 0; kp < k; ++kp) {
        int row = sidx[((bb * NCn + kp) << 10) + n0 + i];
        acc += Gt[(((size_t)(pbase + kp)) << 18) + ((size_t)row << 9) + d];
      }
      Bacc[i * BA_PITCH + d] = acc;
    }
  }
  __syncthreads();

  // ---- stage-1 accumulators, seeded with Bacc ----
  v8f c[2][4];
#pragma unroll
  for (int tm = 0; tm < 2; ++tm)
#pragma unroll
    for (int tn = 0; tn < 4; ++tn)
#pragma unroll
      for (int r = 0; r < 8; ++r)
        c[tm][tn][r] = Bacc[(tm * 16 + r + lhalf * 8) * BA_PITCH + wv * 64 + tn * 16 + l16];

  const int abase  = lhalf ? 16 : 0;    // A frag: 16 consecutive K
  const int abase2 = abase + 8;
  const int bbase  = lhalf ? 16 : 0;    // B frag: 16 consecutive K

  // ---- stage 1: K = 512 in 16 chunks of 32, async double-buffered LDS ----
  auto load_s1 = [&](int kc, int buf) {
    if (tid < 128) {                         // X: 32 rows x 32 cols
      int row = tid >> 2, seg = (tid & 3) * 8;
      async_ld16(XS_OFF + buf * (32 * XS_PITCH * 2) + row * (XS_PITCH * 2) + seg * 2,
                 Xbf + (size_t)(tok0 + row) * Dn + kc * 32 + seg);
    }
    const unsigned short* wg = Wbf + ((size_t)k << 18) + kc * 32;
#pragma unroll
    for (int it = 0; it < 4; ++it) {         // W: 256 rows x 32 cols
      int row = (tid >> 2) + it * 64, seg = (tid & 3) * 8;
      async_ld16(WS_OFF + buf * (512 * WS_PITCH * 2) + row * (WS_PITCH * 2) + seg * 2,
                 wg + (size_t)row * Dn + seg);
    }
  };

  load_s1(0, 0);
  async_wait0();
  __syncthreads();
#pragma unroll 2
  for (int kc = 0; kc < 16; ++kc) {
    int buf = kc & 1;
    if (kc < 15) load_s1(kc + 1, buf ^ 1);
    Frag a[2], b[4];
#pragma unroll
    for (int tm = 0; tm < 2; ++tm) {
      const unsigned short* base = Xs + buf * 32 * XS_PITCH + (tm * 16 + l16) * XS_PITCH;
      a[tm].q[0] = *(const uint4*)(base + abase);
      a[tm].q[1] = *(const uint4*)(base + abase2);
    }
#pragma unroll
    for (int tn = 0; tn < 4; ++tn) {
      const unsigned short* base =
          Ws + buf * 512 * WS_PITCH + (wv * 64 + tn * 16 + l16) * WS_PITCH;
      b[tn].q[0] = *(const uint4*)(base + bbase);
      b[tn].q[1] = *(const uint4*)(base + bbase + 8);
    }
#pragma unroll
    for (int tm = 0; tm < 2; ++tm)
#pragma unroll
      for (int tn = 0; tn < 4; ++tn)
        c[tm][tn] = __builtin_amdgcn_wmma_f32_16x16x32_bf16(
            false, a[tm].v, false, b[tn].v, (short)0, c[tm][tn], false, false);
    async_wait0();
    __syncthreads();
  }

  // ---- relu + bf16 into shared Us[32][512] ----
#pragma unroll
  for (int tm = 0; tm < 2; ++tm)
#pragma unroll
    for (int tn = 0; tn < 4; ++tn)
#pragma unroll
      for (int r = 0; r < 8; ++r) {
        float x = c[tm][tn][r];
        x = x > 0.f ? x : 0.f;
        Us[(tm * 16 + r + lhalf * 8) * US_PITCH + wv * 64 + tn * 16 + l16] = f2bf(x);
      }
  __syncthreads();

  // ---- stage-2 accumulators = P_b broadcast ----
#pragma unroll
  for (int tn = 0; tn < 4; ++tn) {
    float pb = Pb[wv * 64 + tn * 16 + l16];
#pragma unroll
    for (int tm = 0; tm < 2; ++tm)
#pragma unroll
      for (int r = 0; r < 8; ++r)
        c[tm][tn][r] = pb;
  }

  auto load_s2 = [&](int dc, int buf) {
    const unsigned short* pg = Pbf + dc * 32;
#pragma unroll
    for (int it = 0; it < 4; ++it) {
      int row = (tid >> 2) + it * 64, seg = (tid & 3) * 8;
      async_ld16(WS_OFF + buf * (512 * WS_PITCH * 2) + row * (WS_PITCH * 2) + seg * 2,
                 pg + (size_t)row * Dn + seg);
    }
  };

  load_s2(0, 0);
  async_wait0();
  __syncthreads();
#pragma unroll 2
  for (int dc = 0; dc < 16; ++dc) {
    int buf = dc & 1;
    if (dc < 15) load_s2(dc + 1, buf ^ 1);
    Frag a[2], b[4];
#pragma unroll
    for (int tm = 0; tm < 2; ++tm) {
      const unsigned short* base = Us + (tm * 16 + l16) * US_PITCH + dc * 32;
      a[tm].q[0] = *(const uint4*)(base + abase);
      a[tm].q[1] = *(const uint4*)(base + abase2);
    }
#pragma unroll
    for (int tn = 0; tn < 4; ++tn) {
      const unsigned short* base =
          Ws + buf * 512 * WS_PITCH + (wv * 64 + tn * 16 + l16) * WS_PITCH;
      b[tn].q[0] = *(const uint4*)(base + bbase);
      b[tn].q[1] = *(const uint4*)(base + bbase + 8);
    }
#pragma unroll
    for (int tm = 0; tm < 2; ++tm)
#pragma unroll
      for (int tn = 0; tn < 4; ++tn)
        c[tm][tn] = __builtin_amdgcn_wmma_f32_16x16x32_bf16(
            false, a[tm].v, false, b[tn].v, (short)0, c[tm][tn], false, false);
    async_wait0();
    __syncthreads();
  }

  // ---- epilogue: out[k][b][v][n]; C-layout rows are contiguous in n ----
  float* outk = out + (((size_t)k * 16 + bb) << 19);   // 512*1024 floats
#pragma unroll
  for (int tm = 0; tm < 2; ++tm)
#pragma unroll
    for (int tn = 0; tn < 4; ++tn) {
      int vcol = wv * 64 + tn * 16 + l16;
      int nrow = n0 + tm * 16 + lhalf * 8;
      float* dst = outk + ((size_t)vcol << 10) + nrow;
      float4 lo = make_float4(c[tm][tn][0], c[tm][tn][1], c[tm][tn][2], c[tm][tn][3]);
      float4 hi = make_float4(c[tm][tn][4], c[tm][tn][5], c[tm][tn][6], c[tm][tn][7]);
      *(float4*)(dst)     = lo;
      *(float4*)(dst + 4) = hi;
    }
}

// ---------------------------------------------------------------------------
// Host launcher
// ---------------------------------------------------------------------------
extern "C" void kernel_launch(void* const* d_in, const int* in_sizes, int n_in,
                              void* d_out, int out_size, void* d_ws, size_t ws_size,
                              hipStream_t stream) {
  (void)in_sizes; (void)n_in; (void)out_size; (void)ws_size;
  const int*   sidx = (const int*)d_in[0];
  const float* yl   = (const float*)d_in[1];
  const float* lng  = (const float*)d_in[2];
  const float* lnb  = (const float*)d_in[3];
  const float* U0w  = (const float*)d_in[4];
  const float* U0b  = (const float*)d_in[5];
  const float* U1w  = (const float*)d_in[6];
  const float* U1b  = (const float*)d_in[7];
  const float* U2w  = (const float*)d_in[8];
  const float* U2b  = (const float*)d_in[9];
  const float* U3w  = (const float*)d_in[10];
  const float* U3b  = (const float*)d_in[11];
  const float* Pw   = (const float*)d_in[12];
  const float* Pb   = (const float*)d_in[13];
  float* out = (float*)d_out;

  char* ws = (char*)d_ws;
  unsigned short* Xbf = (unsigned short*)ws;                         // 16 MB
  unsigned short* Wbf = (unsigned short*)(ws + 16777216);            // 2 MB
  unsigned short* Pbf = (unsigned short*)(ws + 16777216 + 2097152);  // 0.5 MB
  float*          Gt  = (float*)(ws + 16777216 + 2097152 + 524288);  // 6 MB

  ln_cast_kernel<<<TOKS / 64, 256, 64 * LN_PITCH * 4, stream>>>(yl, lng, lnb, Xbf);
  gather_transpose_kernel<<<(6 * 512 * 512) / 256, 256, 0, stream>>>(U1w, U2w, U3w, Gt);
  wcast_kernel<<<(5 * 512 * 512) / 256, 256, 0, stream>>>(U0w, U1w, U2w, U3w, Pw, Wbf, Pbf);
  fused_gemm_kernel<<<dim3(TOKS / 32, NCn), 256, SM_TOT, stream>>>(
      Xbf, Wbf, Pbf, Gt, sidx, U0b, U1b, U2b, U3b, Pb, out);
}
